// SegmentationAttentionModule_31954556682488
// MI455X (gfx1250) — matrix operand order repaired and checked by reading, not compiled, not run on hardware
//
#include <hip/hip_runtime.h>

// ---------------------------------------------------------------------------
// Problem constants (from reference setup_inputs)
// ---------------------------------------------------------------------------
#define B_    2
#define DK    128
#define DV    512
#define NQ    4096            // H*W
#define NM    8192            // T*H*W
#define QT    32              // queries per block (2 x 16-wide WMMA tiles)
#define PSCAL 40.0f
#define NEGINF (-1e9f)
#define NCH   (NM / 32)       // 256 chunks of 32 memory rows

typedef __attribute__((ext_vector_type(16))) __bf16       v16bf;
typedef __attribute__((ext_vector_type(8)))  float        v8f;
typedef __attribute__((ext_vector_type(4)))  unsigned int u32x4;

union Frag {
    v16bf v;
    u32x4 u[2];
};

// fp32 -> bf16 (round to nearest even), stored as raw u16
static __device__ __forceinline__ unsigned short f2bf(float x) {
    unsigned int u = __float_as_uint(x);
    u += 0x7FFFu + ((u >> 16) & 1u);
    return (unsigned short)(u >> 16);
}

// ---------------------------------------------------------------------------
// Prepass: layout conversions to bf16 in workspace
// ---------------------------------------------------------------------------
__global__ void cvt_q_kernel(const float* __restrict__ qk, unsigned short* __restrict__ Qt) {
    // Qt[b][q][d] = bf16(qkey[b][d][q])
    int i = blockIdx.x * blockDim.x + threadIdx.x;
    if (i >= B_ * NQ * DK) return;
    int d = i % DK;
    int q = (i / DK) % NQ;
    int b = i / (DK * NQ);
    Qt[i] = f2bf(qk[((size_t)b * DK + d) * NQ + q]);
}

__global__ void cvt_k_kernel(const float* __restrict__ mk, unsigned short* __restrict__ Kt) {
    // Kt[b][m][d] = bf16(mkey[b][d][m])
    int i = blockIdx.x * blockDim.x + threadIdx.x;
    if (i >= B_ * NM * DK) return;
    int d = i % DK;
    int m = (i / DK) % NM;
    int b = i / (DK * NM);
    Kt[i] = f2bf(mk[((size_t)b * DK + d) * NM + m]);
}

__global__ void cvt_v_kernel(const float* __restrict__ mv, unsigned short* __restrict__ Vh) {
    // Vh[b][v][m] = bf16(mval[b][v][m])  (layout unchanged, m contiguous)
    int i = blockIdx.x * blockDim.x + threadIdx.x;
    if (i >= B_ * DV * NM) return;
    Vh[i] = f2bf(mv[i]);
}

__global__ void flags_kernel(const unsigned char* __restrict__ qm,
                             const unsigned char* __restrict__ mm,
                             int* __restrict__ flags) {
    __shared__ int sq, sm;
    int b = blockIdx.x;
    if (threadIdx.x == 0) { sq = 0; sm = 0; }
    __syncthreads();
    int aq = 0, am = 0;
    for (int i = threadIdx.x; i < NQ; i += blockDim.x) aq |= qm[(size_t)b * NQ + i];
    for (int i = threadIdx.x; i < NM; i += blockDim.x) am |= mm[(size_t)b * NM + i];
    if (aq) atomicOr(&sq, 1);
    if (am) atomicOr(&sm, 1);
    __syncthreads();
    if (threadIdx.x == 0) { flags[2 * b] = sq; flags[2 * b + 1] = sm; }
}

// ---------------------------------------------------------------------------
// Flash-attention kernel: 5 waves/block. Wave 4 = S/softmax producer
// (32m x 32q per chunk, 16 WMMA). Waves 0..3 = PV consumers (Dv slice of
// 128 each, 16 WMMA). Block = (batch, 32-query tile).
// ---------------------------------------------------------------------------
__launch_bounds__(160)
__global__ void attn_kernel(const unsigned short* __restrict__ Qt,
                            const unsigned short* __restrict__ Kt,
                            const unsigned short* __restrict__ Vh,
                            const int* __restrict__ flags,
                            const unsigned char* __restrict__ qmask,
                            const unsigned char* __restrict__ mmask,
                            float* __restrict__ out) {
    __shared__ __align__(16) unsigned short Pbuf[2][QT][32]; // [buf][q][m_local] bf16
    __shared__ float alphab[2][QT];                          // [buf][q]
    __shared__ float fscale[QT];                             // final 1/l (or 0) per q
    __shared__ __align__(4) unsigned char maskb[32];         // chunk mmask bytes

    const int lane = threadIdx.x & 31;
    const int wave = threadIdx.x >> 5;
    const int b    = blockIdx.y;
    const int q0   = blockIdx.x * QT;
    const int lq   = lane & 15;   // N index within a 16-wide tile
    const int hi   = lane >> 4;   // half-wave selector in A/B/D layouts

    if (wave == 4) {
        // =============== S / softmax producer wave =====================
        // Q as WMMA B-matrices (32d x 16q), 2 q-subtiles x 4 k-steps, in VGPRs
        Frag bq[2][4];
        #pragma unroll
        for (int qt = 0; qt < 2; ++qt) {
            const unsigned short* Qrow = Qt + ((size_t)b * NQ + q0 + qt * 16 + lq) * DK;
            #pragma unroll
            for (int ks = 0; ks < 4; ++ks) {
                const unsigned short* p = Qrow + ks * 32 + hi * 8;
                bq[qt][ks].u[0] = *(const u32x4*)(p);
                bq[qt][ks].u[1] = *(const u32x4*)(p + 16);
            }
        }

        const unsigned short* Kbase = Kt + (size_t)b * NM * DK;
        float mrun[2] = {-3.0e38f, -3.0e38f};
        float lrun[2] = {0.0f, 0.0f};

        #pragma unroll 1
        for (int c = 0; c < NCH; ++c) {
            const int m0 = c * 32;

            // async global->LDS copy of this chunk's 32 mmask bytes (CDNA5 path)
            {
                unsigned           ldsoff = (unsigned)(uintptr_t)(&maskb[lane]);
                unsigned long long ga     =
                    (unsigned long long)(uintptr_t)(mmask + (size_t)b * NM + m0 + lane);
                asm volatile("global_load_async_to_lds_b8 %0, %1, off"
                             :: "v"(ldsoff), "v"(ga) : "memory");
            }

            // S tiles: acc[qt][mt]; mt0 = rows m0..m0+15, mt1 = m0+16..m0+31
            v8f acc[2][2];
            #pragma unroll
            for (int qt = 0; qt < 2; ++qt)
                #pragma unroll
                for (int mt = 0; mt < 2; ++mt)
                    acc[qt][mt] = (v8f){0, 0, 0, 0, 0, 0, 0, 0};

            #pragma unroll
            for (int ks = 0; ks < 4; ++ks) {
                Frag a0, a1;
                const unsigned short* k0 = Kbase + (size_t)(m0 + lq) * DK + ks * 32 + hi * 8;
                const unsigned short* k1 = k0 + 16 * DK;
                a0.u[0] = *(const u32x4*)(k0);
                a0.u[1] = *(const u32x4*)(k0 + 16);
                a1.u[0] = *(const u32x4*)(k1);
                a1.u[1] = *(const u32x4*)(k1 + 16);
                __builtin_prefetch(k0 + 32 * DK, 0, 1);   // next chunk's K rows
                acc[0][0] = __builtin_amdgcn_wmma_f32_16x16x32_bf16(false, a0.v, false, bq[0][ks].v,
                                                                    (short)0, acc[0][0], false, false);
                acc[0][1] = __builtin_amdgcn_wmma_f32_16x16x32_bf16(false, a1.v, false, bq[0][ks].v,
                                                                    (short)0, acc[0][1], false, false);
                acc[1][0] = __builtin_amdgcn_wmma_f32_16x16x32_bf16(false, a0.v, false, bq[1][ks].v,
                                                                    (short)0, acc[1][0], false, false);
                acc[1][1] = __builtin_amdgcn_wmma_f32_16x16x32_bf16(false, a1.v, false, bq[1][ks].v,
                                                                    (short)0, acc[1][1], false, false);
            }

            asm volatile("s_wait_asynccnt 0" ::: "memory");  // mask bytes resident

            #pragma unroll
            for (int qt = 0; qt < 2; ++qt) {
                // scale + mask -> per-lane 16 logits for this lane's query column
                float s[16];
                #pragma unroll
                for (int r = 0; r < 8; ++r) {
                    const int ml0 = r + hi * 8;        // m_local of tile mt0, VGPR r
                    const int ml1 = 16 + r + hi * 8;   // m_local of tile mt1
                    s[r]     = (maskb[ml0] != 0) ? acc[qt][0][r] * PSCAL : NEGINF;
                    s[8 + r] = (maskb[ml1] != 0) ? acc[qt][1][r] * PSCAL : NEGINF;
                }

                // online softmax update (lanes l and l^16 share the same query)
                float tmax = s[0];
                #pragma unroll
                for (int r = 1; r < 16; ++r) tmax = fmaxf(tmax, s[r]);
                tmax = fmaxf(tmax, __shfl_xor(tmax, 16));

                const float mnew  = fmaxf(mrun[qt], tmax);
                const float alpha = __expf(mrun[qt] - mnew);

                float csum = 0.0f;
                unsigned short* Prow = &Pbuf[c & 1][qt * 16 + lq][0];
                #pragma unroll
                for (int r = 0; r < 16; ++r) {
                    const float p = __expf(s[r] - mnew);
                    csum += p;
                    const int ml = (r < 8) ? (r + hi * 8) : (8 + r + hi * 8);
                    Prow[ml] = f2bf(p);
                }
                csum += __shfl_xor(csum, 16);
                lrun[qt] = lrun[qt] * alpha + csum;
                mrun[qt] = mnew;

                if (lane < 16) alphab[c & 1][qt * 16 + lane] = alpha;
            }
            __syncthreads();   // publish P/alpha for chunk c; release buf (c-1)&1
        }

        // epilogue: final normalization factor per query
        const int ok = flags[2 * b] & flags[2 * b + 1];
        if (lane < 16) {
            #pragma unroll
            for (int qt = 0; qt < 2; ++qt) {
                const float qmv = (float)qmask[(size_t)b * NQ + q0 + qt * 16 + lane];
                fscale[qt * 16 + lane] = (ok && qmv != 0.0f) ? (1.0f / lrun[qt]) : 0.0f;
            }
        }
        __syncthreads();       // barrier #NCH: publish fscale
    } else {
        // =============== PV consumer waves (Dv slice of 128) ============
        const int v0 = wave * 128;
        v8f acc[8][2];
        #pragma unroll
        for (int vt = 0; vt < 8; ++vt)
            #pragma unroll
            for (int qt = 0; qt < 2; ++qt)
                acc[vt][qt] = (v8f){0, 0, 0, 0, 0, 0, 0, 0};

        const unsigned short* Vbase = Vh + (size_t)b * DV * NM;

        #pragma unroll 1
        for (int it = 0; it <= NCH; ++it) {
            if (it > 0) {
                const int j  = it - 1;
                const int m0 = j * 32;

                const float a0 = alphab[j & 1][lq];
                const float a1 = alphab[j & 1][16 + lq];

                // P as WMMA B-matrices (32m x 16q) from LDS, one per q-subtile
                Frag pb0, pb1;
                {
                    const unsigned short* p0 = &Pbuf[j & 1][lq][hi * 8];
                    const unsigned short* p1 = &Pbuf[j & 1][16 + lq][hi * 8];
                    pb0.u[0] = *(const u32x4*)(p0);
                    pb0.u[1] = *(const u32x4*)(p0 + 16);
                    pb1.u[0] = *(const u32x4*)(p1);
                    pb1.u[1] = *(const u32x4*)(p1 + 16);
                }

                #pragma unroll
                for (int vt = 0; vt < 8; ++vt) {
                    #pragma unroll
                    for (int r = 0; r < 8; ++r) {
                        acc[vt][0][r] = acc[vt][0][r] * a0;
                        acc[vt][1][r] = acc[vt][1][r] * a1;
                    }
                    Frag av;  // V tile 16v x 32m as A-matrix (reused for both q-subtiles)
                    const unsigned short* vr =
                        Vbase + (size_t)(v0 + vt * 16 + lq) * NM + m0 + hi * 8;
                    av.u[0] = *(const u32x4*)(vr);
                    av.u[1] = *(const u32x4*)(vr + 16);
                    __builtin_prefetch(vr + 64, 0, 1);   // V rows two chunks ahead

                    acc[vt][0] = __builtin_amdgcn_wmma_f32_16x16x32_bf16(false, av.v, false, pb0.v,
                                                                         (short)0, acc[vt][0], false, false);
                    acc[vt][1] = __builtin_amdgcn_wmma_f32_16x16x32_bf16(false, av.v, false, pb1.v,
                                                                         (short)0, acc[vt][1], false, false);
                }
            }
            __syncthreads();   // matches producer barrier sequence
        }

        // write normalized output: out[b][v][q]
        #pragma unroll
        for (int vt = 0; vt < 8; ++vt) {
            const int vbase = v0 + vt * 16 + hi * 8;
            #pragma unroll
            for (int qt = 0; qt < 2; ++qt) {
                const float f = fscale[qt * 16 + lq];
                #pragma unroll
                for (int r = 0; r < 8; ++r) {
                    out[((size_t)b * DV + vbase + r) * NQ + q0 + qt * 16 + lq] =
                        acc[vt][qt][r] * f;
                }
            }
        }
    }
}

// ---------------------------------------------------------------------------
// Launch
// ---------------------------------------------------------------------------
extern "C" void kernel_launch(void* const* d_in, const int* in_sizes, int n_in,
                              void* d_out, int out_size, void* d_ws, size_t ws_size,
                              hipStream_t stream) {
    const float*         qkey  = (const float*)d_in[0];
    // d_in[1] = qval (unused by the reference)
    const unsigned char* qmask = (const unsigned char*)d_in[2];
    const float*         mkey  = (const float*)d_in[3];
    const float*         mval  = (const float*)d_in[4];
    const unsigned char* mmask = (const unsigned char*)d_in[5];
    float*               out   = (float*)d_out;

    unsigned short* Qt = (unsigned short*)d_ws;                  // B*NQ*DK bf16
    unsigned short* Kt = Qt + (size_t)B_ * NQ * DK;              // B*NM*DK bf16
    unsigned short* Vh = Kt + (size_t)B_ * NM * DK;              // B*DV*NM bf16
    int*          flags = (int*)(Vh + (size_t)B_ * DV * NM);     // 2 ints per batch

    {
        int n = B_ * NQ * DK;
        cvt_q_kernel<<<(n + 255) / 256, 256, 0, stream>>>(qkey, Qt);
    }
    {
        int n = B_ * NM * DK;
        cvt_k_kernel<<<(n + 255) / 256, 256, 0, stream>>>(mkey, Kt);
    }
    {
        int n = B_ * DV * NM;
        cvt_v_kernel<<<(n + 255) / 256, 256, 0, stream>>>(mval, Vh);
    }
    flags_kernel<<<B_, 256, 0, stream>>>(qmask, mmask, flags);

    dim3 grid(NQ / QT, B_);
    attn_kernel<<<grid, 160, 0, stream>>>(Qt, Kt, Vh, flags, qmask, mmask, out);
}
